// MemoryEfficientTopKProjector_81638738362601
// MI455X (gfx1250) — matrix operand order repaired and checked by reading, not compiled, not run on hardware
//
#include <hip/hip_runtime.h>

typedef __attribute__((ext_vector_type(16))) _Float16 v16h;
typedef __attribute__((ext_vector_type(8)))  _Float16 v8h;
typedef __attribute__((ext_vector_type(8)))  float    v8f;
typedef __attribute__((ext_vector_type(4)))  float    v4f;
typedef unsigned int u32x4 __attribute__((ext_vector_type(4)));
typedef int          i32x8 __attribute__((ext_vector_type(8)));
typedef int          i32x4 __attribute__((ext_vector_type(4)));

#define ROWS  256      // B*M
#define HDIM  4096
#define VOCAB 32000
#define TOPK  64
#define CHUNK 125      // vocab cols per workgroup: 256 * 125 = 32000
#define KSTEPS (HDIM / 32)
#define SLICE_BYTES 32768   // (256 rows x 32 halves x 2B) x {hi,lo}
#define PLANE_BYTES 16384   // one precision plane within a slice

// ---------------------------------------------------------------------------
// Kernel 1: split f32 A into hi/lo f16 so 3 f16 WMMAs reproduce f32 accuracy.
// Ahi and Alo are contiguous (Alo = Ahi + ROWS*HDIM) so one 3-D TDM tile can
// fetch both planes of a K-slice.
// ---------------------------------------------------------------------------
__global__ void pack_a_kernel(const float* __restrict__ A,
                              _Float16* __restrict__ Ahi,
                              _Float16* __restrict__ Alo) {
  int i = blockIdx.x * blockDim.x + threadIdx.x;
  float a = A[i];
  _Float16 h = (_Float16)a;
  Ahi[i] = h;
  Alo[i] = (_Float16)(a - (float)h);
}

// ---------------------------------------------------------------------------
// TDM: DMA one K-slice of the A-pack (tile 32 x 256 x 2 planes, 32 KB) into
// LDS. Descriptor per CDNA5 ISA.
//   dim0 (x): 32 halves along H        (tensor_dim0 = 4096)
//   dim1 (y): 256 rows                 (tensor_dim0_stride = 4096 elements)
//   dim2 (z): 2 planes hi->lo          (tensor_dim1_stride = ROWS*HDIM elems)
// LDS result: [z][row][k'] => hi plane at +0, lo plane at +PLANE_BYTES,
// row stride 64 B.
// ---------------------------------------------------------------------------
__device__ __forceinline__ void tdm_load_slice(const _Float16* gsrc,
                                               unsigned lds_off) {
  unsigned long long ga = (unsigned long long)(size_t)gsrc;
  u32x4 g0;
  g0[0] = 1u;                                   // count=1, user desc
  g0[1] = lds_off;                              // lds_addr (bytes)
  g0[2] = (unsigned)(ga & 0xffffffffu);         // global_addr[31:0]
  g0[3] = (unsigned)((ga >> 32) & 0x01ffffffu)  // global_addr[56:32]
          | (2u << 30);                         // type = 2 (image)
  i32x8 g1;
  g1[0] = (int)(1u << 16);                      // data_size=1 (2 bytes)
  g1[1] = (int)((HDIM & 0xffffu) << 16);        // tensor_dim0[15:0] @bit48
  g1[2] = (int)(((unsigned)HDIM >> 16) | ((ROWS & 0xffffu) << 16)); // dim0 hi | dim1 lo
  g1[3] = (int)(((unsigned)ROWS >> 16) | (32u << 16));              // dim1 hi | tile_dim0=32
  g1[4] = (int)(256u | (2u << 16));             // tile_dim1=256 | tile_dim2=2
  g1[5] = (int)HDIM;                            // tensor_dim0_stride[31:0] = 4096
  g1[6] = (int)(((unsigned)(ROWS * HDIM) & 0xffffu) << 16);  // d0s hi16 | d1s[15:0]
  g1[7] = (int)((unsigned)(ROWS * HDIM) >> 16); // tensor_dim1_stride[47:16]
  i32x4 g2;
  g2[0] = 2;                                    // tensor_dim2 = 2 planes
  g2[1] = 0;                                    // tensor_dim3 unused
  g2[2] = 0;                                    // tensor_dim2_stride unused
  g2[3] = 0;                                    // tile_dim3 = 0 (unused)
  i32x4 g3;
  g3[0] = 0; g3[1] = 0; g3[2] = 0; g3[3] = 0;
  i32x8 g4;                                     // extra operand of the 6-arg
  g4[0] = 0; g4[1] = 0; g4[2] = 0; g4[3] = 0;   // toolchain signature; zeroed
  g4[4] = 0; g4[5] = 0; g4[6] = 0; g4[7] = 0;
  __builtin_amdgcn_tensor_load_to_lds(g0, g1, g2, g3, g4, 0);
}

// ---------------------------------------------------------------------------
// Kernel 2: fused split-precision GEMM. Each workgroup owns a 125-wide vocab
// chunk for ALL 256 rows (W_vocab read exactly once from HBM). 8 waves; wave
// w owns N-tile w; 16 M-tile f32 accumulators per wave. A K-slices are DMAed
// into double-buffered LDS by the TDM and read back with ds_load_b128; the B
// fragment is loaded from global, split hi/lo once per K-step, and reused
// across all 16 M-tiles (3 WMMAs each).
// ---------------------------------------------------------------------------
__global__ __launch_bounds__(256, 1)
void gemm_logits_kernel(const _Float16* __restrict__ Ahi,
                        const float* __restrict__ Wv,
                        float* __restrict__ logits) {
  extern __shared__ char smem[];                // 2 x 32 KB slice buffers
  const unsigned lds_base = (unsigned)(size_t)smem; // LDS byte offset (as(3))

  const int chunk = blockIdx.x;
  const int lane  = threadIdx.x & 31;
  const int wave  = threadIdx.x >> 5;
  const int m     = lane & 15;   // A-row / B-col / C-col within tile
  const int g     = lane >> 4;   // lane half-group
  const int cIn   = wave * 16 + m;          // column within chunk [0,128)
  const bool cok  = cIn < CHUNK;
  const int vRow  = chunk * CHUNK + (cok ? cIn : CHUNK - 1);
  // B fragment: lane (n,g) holds K = g*16 .. g*16+15 of column n
  const float* __restrict__ bbase = Wv + (size_t)vRow * HDIM + g * 16;

  v8f acc[16];
  #pragma unroll
  for (int mt = 0; mt < 16; ++mt) acc[mt] = (v8f)(0.0f);

  // Prologue: DMA slices 0 and 1 (wave 0 only; EXEC-independent, one op each)
  if (wave == 0) {
    tdm_load_slice(Ahi + 0 * 32, lds_base);
    tdm_load_slice(Ahi + 1 * 32, lds_base + SLICE_BYTES);
  }

  #pragma unroll 1
  for (int ks = 0; ks < KSTEPS; ++ks) {
    if (wave == 0) {
      if (ks + 1 < KSTEPS) __builtin_amdgcn_s_wait_tensorcnt(1);
      else                 __builtin_amdgcn_s_wait_tensorcnt(0);
    }
    __syncthreads();   // slice ks resident in LDS for all waves

    // --- B fragment: global load + hi/lo split (once per K-step) ---
    const float* bp = bbase + ks * 32;
    __builtin_prefetch(bp + 32, 0, 1);
    v4f b0 = *(const v4f*)(bp + 0);
    v4f b1 = *(const v4f*)(bp + 4);
    v4f b2 = *(const v4f*)(bp + 8);
    v4f b3 = *(const v4f*)(bp + 12);
    float bf[16];
    #pragma unroll
    for (int i2 = 0; i2 < 4; ++i2) {
      bf[i2] = b0[i2]; bf[4 + i2] = b1[i2]; bf[8 + i2] = b2[i2]; bf[12 + i2] = b3[i2];
    }
    v16h bhi, blo;
    #pragma unroll
    for (int i2 = 0; i2 < 16; ++i2) {
      _Float16 h = (_Float16)bf[i2];
      bhi[i2] = h;
      blo[i2] = (_Float16)(bf[i2] - (float)h);
    }

    // --- A fragments from LDS (ds_load_b128), 3 WMMAs per M-tile ---
    const char* buf = smem + (ks & 1) * SLICE_BYTES;
    #pragma unroll
    for (int mt = 0; mt < 16; ++mt) {
      const int r = mt * 16 + m;
      const char* hp = buf + r * 64 + g * 32;
      const char* lp = hp + PLANE_BYTES;
      v8h h0 = *(const v8h*)(hp);
      v8h h1 = *(const v8h*)(hp + 16);
      v8h l0 = *(const v8h*)(lp);
      v8h l1 = *(const v8h*)(lp + 16);
      v16h ah = __builtin_shufflevector(h0, h1, 0,1,2,3,4,5,6,7,8,9,10,11,12,13,14,15);
      v16h al = __builtin_shufflevector(l0, l1, 0,1,2,3,4,5,6,7,8,9,10,11,12,13,14,15);
      acc[mt] = __builtin_amdgcn_wmma_f32_16x16x32_f16(false, al, false, bhi, (short)0, acc[mt], false, false);
      acc[mt] = __builtin_amdgcn_wmma_f32_16x16x32_f16(false, ah, false, blo, (short)0, acc[mt], false, false);
      acc[mt] = __builtin_amdgcn_wmma_f32_16x16x32_f16(false, ah, false, bhi, (short)0, acc[mt], false, false);
    }

    __syncthreads();   // all waves done with buffer (ks & 1)
    if (wave == 0 && ks + 2 < KSTEPS) {
      tdm_load_slice(Ahi + (size_t)(ks + 2) * 32,
                     lds_base + (unsigned)((ks & 1) * SLICE_BYTES));
    }
  }

  if (cok) {
    const int col = chunk * CHUNK + cIn;
    #pragma unroll
    for (int mt = 0; mt < 16; ++mt) {
      #pragma unroll
      for (int j = 0; j < 8; ++j) {
        int r = mt * 16 + g * 8 + j;   // C layout: VGPR j -> M = j + 8*g
        logits[(size_t)r * VOCAB + col] = acc[mt][j];
      }
    }
  }
}

// ---------------------------------------------------------------------------
// Kernel 3: per-row top-64 (exact, index-tiebreak like jax top_k) + softmax
// + gather-weighted embedding sum. One workgroup per row; logit row staged in
// LDS (125 KB of the 320 KB CDNA5 LDS).
// ---------------------------------------------------------------------------
__global__ __launch_bounds__(256)
void topk_softmax_gather_kernel(const float* __restrict__ logits,
                                const float* __restrict__ embed,
                                const float* __restrict__ tau_p,
                                float* __restrict__ out) {
  const int row = blockIdx.x;
  const int t   = threadIdx.x;
  extern __shared__ float fsmem[];
  float* vals = fsmem;                 // 32000
  float* sv   = fsmem + 32000;         // 256
  int*   si   = (int*)(fsmem + 32256); // 256
  float* topv = fsmem + 32512;         // 64
  int*   topi = (int*)(fsmem + 32576); // 64
  float* pbuf = fsmem + 32640;         // 64   (total 32704 floats)

  const float* lrow = logits + (size_t)row * VOCAB;
  for (int j = 0; j < VOCAB / 256; ++j) vals[j * 256 + t] = lrow[j * 256 + t];
  __syncthreads();

  for (int r = 0; r < TOPK; ++r) {
    float bv = -1e30f; int bi = 0x7fffffff;
    for (int j = 0; j < VOCAB / 256; ++j) {
      int i = j * 256 + t;
      float v = vals[i];
      if (v > bv) { bv = v; bi = i; }     // ascending i => smallest index on tie
    }
    sv[t] = bv; si[t] = bi;
    __syncthreads();
    for (int s = 128; s > 0; s >>= 1) {
      if (t < s) {
        float rv = sv[t + s]; int ri = si[t + s];
        if (rv > sv[t] || (rv == sv[t] && ri < si[t])) { sv[t] = rv; si[t] = ri; }
      }
      __syncthreads();
    }
    if (t == 0) { topv[r] = sv[0]; topi[r] = si[0]; vals[si[0]] = -1e30f; }
    __syncthreads();
  }

  if (t == 0) {
    float tau = tau_p[0];
    float mx = topv[0];                  // round 0 is the global max
    float s = 0.f;
    for (int r = 0; r < TOPK; ++r) { float e = __expf((topv[r] - mx) / tau); pbuf[r] = e; s += e; }
    float inv = 1.f / s;
    for (int r = 0; r < TOPK; ++r) pbuf[r] *= inv;
  }
  __syncthreads();

  float acc[HDIM / 256];
  #pragma unroll
  for (int j = 0; j < HDIM / 256; ++j) acc[j] = 0.f;
  for (int k = 0; k < TOPK; ++k) {
    float w = pbuf[k];
    const float* e = embed + (size_t)topi[k] * HDIM;
    #pragma unroll
    for (int j = 0; j < HDIM / 256; ++j) acc[j] += w * e[t + j * 256];
  }
  float* orow = out + (size_t)row * HDIM;
  #pragma unroll
  for (int j = 0; j < HDIM / 256; ++j) orow[t + j * 256] = acc[j];

  if (t < TOPK) {
    out[(size_t)ROWS * HDIM + (size_t)row * TOPK + t] = (float)topi[t];
  }
}

// ---------------------------------------------------------------------------
extern "C" void kernel_launch(void* const* d_in, const int* in_sizes, int n_in,
                              void* d_out, int out_size, void* d_ws, size_t ws_size,
                              hipStream_t stream) {
  const float* prefix = (const float*)d_in[0];  // [256, 4096] f32
  const float* embed  = (const float*)d_in[1];  // [32000, 4096] f32
  const float* Wvocab = (const float*)d_in[2];  // [32000, 4096] f32
  const float* tau    = (const float*)d_in[3];  // scalar f32
  // d_in[4] = k (== 64, compiled in as TOPK)

  // workspace layout: Ahi (2 MB) | Alo (2 MB, contiguous) | logits (32.77 MB)
  _Float16* Ahi = (_Float16*)d_ws;
  _Float16* Alo = Ahi + (size_t)ROWS * HDIM;
  float* logits = (float*)((char*)d_ws + 4ull * 1024 * 1024);
  float* out    = (float*)d_out;

  pack_a_kernel<<<(ROWS * HDIM) / 256, 256, 0, stream>>>(prefix, Ahi, Alo);
  gemm_logits_kernel<<<VOCAB / CHUNK, 256, 2 * SLICE_BYTES, stream>>>(
      Ahi, Wvocab, logits);
  topk_softmax_gather_kernel<<<ROWS, 256, 32704 * sizeof(float), stream>>>(
      logits, embed, tau, out);
}
